// MemModule_78572131713360
// MI455X (gfx1250) — matrix operand order repaired and checked by reading, compiled, not run on hardware
//
#include <hip/hip_runtime.h>
#include <cstdint>
#include <cstddef>

// ---------------- problem constants ----------------
#define MEM_DIM   2000
#define MEM_PAD   2048          // padded to 128 N-tiles of 16
#define MEM_TILES_VALID 125     // 2000/16
#define FEA_DIM   512
#define N_TOK     32768         // 32 * 32 * 32
#define HW        1024          // 32*32
#define SHRINK    0.0025f
#define EPSV      1e-12f

// ---------------- CDNA5 WMMA types ----------------
typedef __attribute__((ext_vector_type(16))) __bf16        v16bf;
typedef __attribute__((ext_vector_type(8)))  float         v8f;
typedef __attribute__((ext_vector_type(4)))  unsigned int  u32x4;

union FragAB {           // 16 bf16 = 8 VGPRs, loadable as 2x b128
    v16bf v;
    u32x4 q[2];
};

__device__ __forceinline__ unsigned short f2bf(float f) {
    union { float f; unsigned u; } x; x.f = f;
    unsigned r = x.u + 0x7FFFu + ((x.u >> 16) & 1u);   // round-to-nearest-even
    return (unsigned short)(r >> 16);
}

// ---- CDNA5 async global->LDS copy (ASYNCcnt tracked), §4 of CDNA5 ISA ----
// vdst operand = per-lane LDS byte offset; vaddr = 64-bit global address.
__device__ __forceinline__ void async_load_b128(const void* gaddr, unsigned lds_off) {
    asm volatile("global_load_async_to_lds_b128 %0, %1, off"
                 :: "v"(lds_off), "v"(gaddr) : "memory");
}
__device__ __forceinline__ void wait_async0() {
    asm volatile("s_wait_asynccnt 0" ::: "memory");
}
__device__ __forceinline__ unsigned lds_off32(const void* p) {
    return (unsigned)(uintptr_t)p;     // low 32 bits of generic ptr == LDS offset
}

// ---------------- prep kernels ----------------
// NCHW fp32 -> token-major bf16  xf[n][c],  n = b*1024 + h*32 + w
__global__ void prep_x_kernel(const float* __restrict__ x,
                              unsigned short* __restrict__ xf) {
    int e = blockIdx.x * blockDim.x + threadIdx.x;      // n*512 + c
    if (e >= N_TOK * FEA_DIM) return;
    int n = e >> 9;
    int c = e & 511;
    int b = n >> 10;
    int hw = n & 1023;
    xf[e] = f2bf(x[(size_t)b * FEA_DIM * HW + (size_t)c * HW + hw]);
}

// weight fp32 (2000x512) -> bf16 wB[2048][512] (zero-pad rows) and wT[512][2048]
__global__ void prep_w_kernel(const float* __restrict__ w,
                              unsigned short* __restrict__ wB,
                              unsigned short* __restrict__ wT) {
    int e = blockIdx.x * blockDim.x + threadIdx.x;      // m*512 + c
    if (e >= MEM_PAD * FEA_DIM) return;
    int m = e >> 9;
    int c = e & 511;
    unsigned short v = 0;
    if (m < MEM_DIM) v = f2bf(w[m * FEA_DIM + c]);
    wB[e] = v;
    wT[(size_t)c * MEM_PAD + m] = v;
}

// ---------------- GEMM1 + softmax + shrink (fused) ----------------
// block = 256 threads (8 waves), tile = 16 tokens x 2048 cols, K = 512
__launch_bounds__(256)
__global__ void gemm1_softmax_kernel(const unsigned short* __restrict__ xf,
                                     const unsigned short* __restrict__ wB,
                                     unsigned short* __restrict__ attB,
                                     float* __restrict__ attOut) {
    __shared__ __attribute__((aligned(16))) unsigned short sA[16 * FEA_DIM]; // 16KB
    __shared__ float sRed[8 * 16];

    const int tid  = threadIdx.x;
    const int wave = tid >> 5;
    const int lane = tid & 31;
    const int l    = lane & 15;           // N-col / M-row within tile
    const int g    = lane >> 4;           // half-wave group
    const int tok0 = blockIdx.x * 16;

    // async stage of the 16x512 bf16 A tile (64B per thread, direct to LDS)
    {
        int row = tid >> 4;               // 0..15
        int seg = tid & 15;               // 32 elems (64B) each
        const unsigned short* src = xf + (size_t)(tok0 + row) * FEA_DIM + seg * 32;
        unsigned dlo = lds_off32(sA + row * FEA_DIM + seg * 32);
        async_load_b128(src,      dlo);
        async_load_b128(src + 8,  dlo + 16);
        async_load_b128(src + 16, dlo + 32);
        async_load_b128(src + 24, dlo + 48);
        wait_async0();
    }
    __syncthreads();

    v8f acc[16];
    const v8f vzero = {0.f,0.f,0.f,0.f,0.f,0.f,0.f,0.f};
#pragma unroll
    for (int t = 0; t < 16; ++t) acc[t] = vzero;

    const int tbase = wave * 16;          // this wave owns N-tiles tbase..tbase+15

    for (int kk = 0; kk < FEA_DIM / 32; ++kk) {
        // A fragment: lane row m=l; k = kk*32 + 16g + i
        FragAB a;
        const unsigned short* ap = sA + l * FEA_DIM + kk * 32 + g * 16;
        a.q[0] = *(const u32x4*)(ap);          // k = 16g .. 16g+7
        a.q[1] = *(const u32x4*)(ap + 8);      // k = 16g+8 .. 16g+15
#pragma unroll
        for (int t = 0; t < 16; ++t) {
            // B = W^T fragment: lane col n, k = kk*32 + 16g + i (contiguous in W row)
            int col = (tbase + t) * 16 + l;
            const unsigned short* bp = wB + (size_t)col * FEA_DIM + kk * 32 + g * 16;
            FragAB b;
            b.q[0] = *(const u32x4*)(bp);
            b.q[1] = *(const u32x4*)(bp + 8);
            acc[t] = __builtin_amdgcn_wmma_f32_16x16x32_bf16(
                false, a.v, false, b.v, (short)0, acc[t], false, false);
        }
    }

    // ---- softmax over 2000 cols; row m = r + 8g, this wave holds 256 cols ----
    // 1) row max
    float rmax[8];
#pragma unroll
    for (int r = 0; r < 8; ++r) {
        float p = -3.0e38f;
#pragma unroll
        for (int t = 0; t < 16; ++t)
            if (tbase + t < MEM_TILES_VALID) p = fmaxf(p, acc[t][r]);
        for (int m = 1; m < 16; m <<= 1) p = fmaxf(p, __shfl_xor(p, m, 32));
        rmax[r] = p;
    }
    __syncthreads();
    if (l == 0) {
#pragma unroll
        for (int r = 0; r < 8; ++r) sRed[wave * 16 + g * 8 + r] = rmax[r];
    }
    __syncthreads();
#pragma unroll
    for (int r = 0; r < 8; ++r) {
        float v = -3.0e38f;
#pragma unroll
        for (int w = 0; w < 8; ++w) v = fmaxf(v, sRed[w * 16 + g * 8 + r]);
        rmax[r] = v;
    }

    // 2) exp + row sum
    float rsum[8];
#pragma unroll
    for (int r = 0; r < 8; ++r) {
        float p = 0.f;
#pragma unroll
        for (int t = 0; t < 16; ++t) {
            float e = (tbase + t < MEM_TILES_VALID) ? __expf(acc[t][r] - rmax[r]) : 0.f;
            acc[t][r] = e;
            p += e;
        }
        for (int m = 1; m < 16; m <<= 1) p += __shfl_xor(p, m, 32);
        rsum[r] = p;
    }
    __syncthreads();
    if (l == 0) {
#pragma unroll
        for (int r = 0; r < 8; ++r) sRed[wave * 16 + g * 8 + r] = rsum[r];
    }
    __syncthreads();
#pragma unroll
    for (int r = 0; r < 8; ++r) {
        float v = 0.f;
#pragma unroll
        for (int w = 0; w < 8; ++w) v += sRed[w * 16 + g * 8 + r];
        rsum[r] = v;
    }

    // 3) hard-shrink + L1 norm
    float rnrm[8];
#pragma unroll
    for (int r = 0; r < 8; ++r) {
        float inv = 1.0f / rsum[r];
        float p = 0.f;
#pragma unroll
        for (int t = 0; t < 16; ++t) {
            float a  = acc[t][r] * inv;
            float d  = a - SHRINK;
            float a2 = fmaxf(d, 0.f) * a / (fabsf(d) + EPSV);
            acc[t][r] = a2;
            p += a2;                        // a2 >= 0 so L1 == sum
        }
        for (int m = 1; m < 16; m <<= 1) p += __shfl_xor(p, m, 32);
        rnrm[r] = p;
    }
    __syncthreads();
    if (l == 0) {
#pragma unroll
        for (int r = 0; r < 8; ++r) sRed[wave * 16 + g * 8 + r] = rnrm[r];
    }
    __syncthreads();
#pragma unroll
    for (int r = 0; r < 8; ++r) {
        float v = 0.f;
#pragma unroll
        for (int w = 0; w < 8; ++w) v += sRed[w * 16 + g * 8 + r];
        rnrm[r] = v;
    }

    // 4) store: bf16 row-major (GEMM2 operand, padded cols -> 0) + fp32 NMHW output
#pragma unroll
    for (int r = 0; r < 8; ++r) {
        float s = 1.0f / fmaxf(rnrm[r], EPSV);
        int n  = tok0 + g * 8 + r;          // token = D-row r+8g
        int b  = n >> 10, hw = n & 1023;
        size_t rowB = (size_t)n * MEM_PAD;
        size_t rowO = (size_t)b * MEM_DIM * HW + hw;
#pragma unroll
        for (int t = 0; t < 16; ++t) {
            int col = (tbase + t) * 16 + l;
            bool valid = (tbase + t) < MEM_TILES_VALID;
            float a = acc[t][r] * s;
            attB[rowB + col] = valid ? f2bf(a) : (unsigned short)0;
            if (valid) attOut[rowO + (size_t)col * HW] = a;
        }
    }
}

// ---------------- GEMM2: y = att @ W ----------------
// block = 256 threads, tile = 16 tokens x 512 feat, K = 2048 in 8x 256-chunks
// A chunks double-buffered in LDS via async global->LDS loads.
__launch_bounds__(256)
__global__ void gemm2_kernel(const unsigned short* __restrict__ attB,
                             const unsigned short* __restrict__ wT,
                             float* __restrict__ yOut) {
    __shared__ __attribute__((aligned(16))) unsigned short sA[2][16 * 256]; // 2x 8KB
    const int tid  = threadIdx.x;
    const int wave = tid >> 5;
    const int lane = tid & 31;
    const int l    = lane & 15;
    const int g    = lane >> 4;
    const int tok0 = blockIdx.x * 16;

    v8f acc[4];
    const v8f vzero = {0.f,0.f,0.f,0.f,0.f,0.f,0.f,0.f};
#pragma unroll
    for (int t = 0; t < 4; ++t) acc[t] = vzero;

    const int tbase = wave * 4;           // 32 N-tiles, 4 per wave
    const int row   = tid >> 4;           // staging: 16 elems (32B) per thread
    const int seg   = tid & 15;

    auto stage = [&](int ch) {
        const unsigned short* src = attB + (size_t)(tok0 + row) * MEM_PAD
                                         + ch * 256 + seg * 16;
        unsigned dlo = lds_off32(&sA[ch & 1][row * 256 + seg * 16]);
        async_load_b128(src,     dlo);
        async_load_b128(src + 8, dlo + 16);
    };

    stage(0);                              // prologue

    for (int ch = 0; ch < MEM_PAD / 256; ++ch) {
        wait_async0();                     // own async writes done
        __syncthreads();                   // all waves' writes visible
        if (ch + 1 < MEM_PAD / 256) stage(ch + 1);   // prefetch next chunk

        const unsigned short* sbuf = sA[ch & 1];
        for (int kk = 0; kk < 8; ++kk) {
            FragAB a;
            const unsigned short* ap = sbuf + l * 256 + kk * 32 + g * 16;
            a.q[0] = *(const u32x4*)(ap);
            a.q[1] = *(const u32x4*)(ap + 8);
            int kglob = ch * 256 + kk * 32;
#pragma unroll
            for (int t = 0; t < 4; ++t) {
                int col = (tbase + t) * 16 + l;        // feature index
                const unsigned short* bp = wT + (size_t)col * MEM_PAD + kglob + g * 16;
                FragAB b;
                b.q[0] = *(const u32x4*)(bp);
                b.q[1] = *(const u32x4*)(bp + 8);
                acc[t] = __builtin_amdgcn_wmma_f32_16x16x32_bf16(
                    false, a.v, false, b.v, (short)0, acc[t], false, false);
            }
        }
    }

    // store y in NCHW: out[b*512*1024 + c*1024 + hw]
#pragma unroll
    for (int r = 0; r < 8; ++r) {
        int n = tok0 + g * 8 + r;
        int b = n >> 10, hw = n & 1023;
        size_t base = (size_t)b * FEA_DIM * HW + hw;
#pragma unroll
        for (int t = 0; t < 4; ++t) {
            int c = (tbase + t) * 16 + l;
            yOut[base + (size_t)c * HW] = acc[t][r];
        }
    }
}

// ---------------- host launcher ----------------
extern "C" void kernel_launch(void* const* d_in, const int* in_sizes, int n_in,
                              void* d_out, int out_size, void* d_ws, size_t ws_size,
                              hipStream_t stream) {
    const float* x = (const float*)d_in[0];     // (32, 512, 32, 32)
    const float* w = (const float*)d_in[1];     // (2000, 512)

    float* outY   = (float*)d_out;                          // 32*512*32*32
    float* outAtt = outY + (size_t)N_TOK * FEA_DIM;         // 32*2000*32*32

    char* ws = (char*)d_ws;
    unsigned short* xf   = (unsigned short*)ws;  ws += (size_t)N_TOK  * FEA_DIM * 2; // 32 MB
    unsigned short* wB   = (unsigned short*)ws;  ws += (size_t)MEM_PAD * FEA_DIM * 2; //  2 MB
    unsigned short* wT   = (unsigned short*)ws;  ws += (size_t)FEA_DIM * MEM_PAD * 2; //  2 MB
    unsigned short* attB = (unsigned short*)ws;  // 32768*2048*2 = 128 MB

    prep_x_kernel<<<(N_TOK * FEA_DIM) / 256, 256, 0, stream>>>(x, xf);
    prep_w_kernel<<<(MEM_PAD * FEA_DIM) / 256, 256, 0, stream>>>(w, wB, wT);
    gemm1_softmax_kernel<<<N_TOK / 16, 256, 0, stream>>>(xf, wB, attB, outAtt);
    gemm2_kernel<<<N_TOK / 16, 256, 0, stream>>>(attB, wT, outY);
}
